// AttentionBlock_21981642621269
// MI455X (gfx1250) — matrix operand order repaired and checked by reading, not compiled, not run on hardware
//
#include <hip/hip_runtime.h>
#include <hip/hip_bf16.h>

// ---------------------------------------------------------------------------
// Types matching the CDNA5 WMMA builtin signatures (probe-confirmed).
// ---------------------------------------------------------------------------
typedef __attribute__((ext_vector_type(16))) __bf16 v16bf;
typedef __attribute__((ext_vector_type(8)))  __bf16 v8bf;
typedef __attribute__((ext_vector_type(4)))  __bf16 v4bf;
typedef __attribute__((ext_vector_type(8)))  float  v8f;

#define BB    8
#define NN    8192
#define DD    512
#define WSZ   128
#define NWIN  64
#define MROWS (BB * NN)          // 65536 total rows (B*N)
#define QSCALE 0.044194173824159216f  // 1/sqrt(512), folded into q

// A-fragment (16x32 bf16, M x K): lane holds row M = lane%16; its 16 values
// are two 8-element K-chunks at kbase+16*(lane/16) and
// kbase+16*(lane/16)+8, i.e. 16 contiguous K (same order as B).
static __device__ __forceinline__ v16bf load_a_frag(const __bf16* p0, const __bf16* p1) {
    v8bf lo = *(const v8bf*)p0;
    v8bf hi = *(const v8bf*)p1;
    v16bf a;
#pragma unroll
    for (int i = 0; i < 8; ++i) { a[i] = lo[i]; a[i + 8] = hi[i]; }
    return a;
}

static __device__ __forceinline__ v8f wmma_bf16(v16bf a, v16bf b, v8f c) {
    // 8 args: (neg_a, A, neg_b, B, c_mod, C, reuse_a, reuse_b)
    return __builtin_amdgcn_wmma_f32_16x16x32_bf16(false, a, false, b, (short)0, c,
                                                   false, false);
}

// ---------------------------------------------------------------------------
// fp32 -> bf16 bulk conversion (count divisible by 4*blockDim*gridDim)
// ---------------------------------------------------------------------------
__global__ void cvt_f32_bf16(const float* __restrict__ in, __bf16* __restrict__ out) {
    int i = (blockIdx.x * blockDim.x + threadIdx.x) * 4;
    float4 f = *(const float4*)(in + i);
    v4bf o;
    o[0] = (__bf16)f.x; o[1] = (__bf16)f.y; o[2] = (__bf16)f.z; o[3] = (__bf16)f.w;
    *(v4bf*)(out + i) = o;
}

// ---------------------------------------------------------------------------
// Fused QKV projection: out[r,c] = sum_d x[r,d]*w[c,d] + b[c]
// blockIdx.z selects projection (0=q with folded scale, 1=k, 2=v transposed).
// Block computes 128(M)x256(N); 8 waves in 2(M)x4(N) grid, each 64x64:
// 16 WMMAs per k-step against 16 b128 fragment loads (1.0 loads/WMMA, vs 1.5
// for a 32x64 tile) -> lower VMEM issue pressure while data sits in WGP$/L2.
// ---------------------------------------------------------------------------
__global__ void __launch_bounds__(256, 1)
qkv_gemm(const __bf16* __restrict__ xb, const __bf16* __restrict__ wb,
         const float* __restrict__ b_q, const float* __restrict__ b_k,
         const float* __restrict__ b_v,
         __bf16* __restrict__ qb, __bf16* __restrict__ kb, __bf16* __restrict__ vT)
{
    const int z    = blockIdx.z;
    const int wave = threadIdx.x >> 5;
    const int lane = threadIdx.x & 31;
    const int half = lane >> 4;
    const int n16  = lane & 15;
    const int mbase = blockIdx.x * 128 + (wave & 1) * 64;
    const int nbase = blockIdx.y * 256 + (wave >> 1) * 64;
    const __bf16* w0 = wb + z * (DD * DD);

    v8f acc[4][4] = {};
#pragma unroll 2
    for (int k = 0; k < DD; k += 32) {
        v16bf a[4];
#pragma unroll
        for (int mt = 0; mt < 4; ++mt) {
            const __bf16* p = xb + (mbase + mt * 16 + n16) * DD + k;
            a[mt] = load_a_frag(p + 16 * half, p + 16 * half + 8);
        }
#pragma unroll
        for (int nt = 0; nt < 4; ++nt) {
            // B-frag (32x16, K x N): lane holds column N = lane%16,
            // 16 contiguous K values starting at k + 16*(lane/16).
            const __bf16* p = w0 + (nbase + nt * 16 + n16) * DD + k + 16 * half;
            v16bf bf = *(const v16bf*)p;
#pragma unroll
            for (int mt = 0; mt < 4; ++mt)
                acc[mt][nt] = wmma_bf16(a[mt], bf, acc[mt][nt]);
        }
    }

    const float* bias = (z == 0) ? b_q : (z == 1) ? b_k : b_v;
#pragma unroll
    for (int nt = 0; nt < 4; ++nt) {
        const int col = nbase + nt * 16 + n16;   // C layout: N = lane%16
        const float bv = bias[col];
#pragma unroll
        for (int mt = 0; mt < 4; ++mt) {
#pragma unroll
            for (int r = 0; r < 8; ++r) {        // C layout: M = r + 8*(lane/16)
                const int row = mbase + mt * 16 + r + 8 * half;
                const float v = acc[mt][nt][r] + bv;
                if (z == 0)      qb[row * DD + col] = (__bf16)(v * QSCALE);
                else if (z == 1) kb[row * DD + col] = (__bf16)v;
                else             vT[col * MROWS + row] = (__bf16)v;  // transposed
            }
        }
    }
}

// ---------------------------------------------------------------------------
// Local attention: one block per (window, batch). 8 waves; wave w owns query
// rows [wstart+16w, wstart+16w+16) against all 256 keys (prev+current window).
// Phase 1: sim via WMMA (16 accum tiles). Mask + softmax (lane-half shuffles).
// Stage unnormalized bf16 weights in wave-private LDS, then phase 2:
// out = attn @ v using vT for contiguous B-fragments; scale by 1/rowsum.
// ---------------------------------------------------------------------------
__global__ void __launch_bounds__(256, 1)
local_attn(const __bf16* __restrict__ qb, const __bf16* __restrict__ kb,
           const __bf16* __restrict__ vT, float* __restrict__ out)
{
    __shared__ __bf16 lds_attn[8 * 16 * 256];   // 64 KB

    const int win  = blockIdx.x;
    const int b    = blockIdx.y;
    const int wave = threadIdx.x >> 5;
    const int lane = threadIdx.x & 31;
    const int half = lane >> 4;
    const int n16  = lane & 15;
    const int wstart = win * WSZ;
    const int qrow0  = b * NN + wstart + wave * 16;

    // ---- Phase 1: sim = q @ k^T over 256 keys (16 key tiles) ----
    v8f acc[16] = {};
#pragma unroll 2
    for (int k = 0; k < DD; k += 32) {
        const __bf16* qp = qb + (qrow0 + n16) * DD + k;
        v16bf a = load_a_frag(qp + 16 * half, qp + 16 * half + 8);
#pragma unroll
        for (int t = 0; t < 16; ++t) {
            int kpos = wstart - WSZ + t * 16 + n16;
            kpos = kpos < 0 ? 0 : kpos;          // clamp; masked to -inf below
            const __bf16* kp = kb + (b * NN + kpos) * DD + k + 16 * half;
            acc[t] = wmma_bf16(a, *(const v16bf*)kp, acc[t]);
        }
    }

    // ---- Mask: causal (tq < tk) and lookback padding (tk < 0) ----
#pragma unroll
    for (int t = 0; t < 16; ++t) {
#pragma unroll
        for (int r = 0; r < 8; ++r) {
            const int tk = wstart - WSZ + t * 16 + n16;
            const int tq = wstart + wave * 16 + r + 8 * half;
            acc[t][r] = (tk < 0 || tq < tk) ? -3.0e38f : acc[t][r];
        }
    }

    // ---- Softmax over 256 keys per row (row = one lane-half of C layout) ----
    float recip[8];
#pragma unroll
    for (int r = 0; r < 8; ++r) {
        float m = acc[0][r];
#pragma unroll
        for (int t = 1; t < 16; ++t) m = fmaxf(m, acc[t][r]);
#pragma unroll
        for (int i = 1; i < 16; i <<= 1) m = fmaxf(m, __shfl_xor(m, i, 32));
        float s = 0.0f;
#pragma unroll
        for (int t = 0; t < 16; ++t) {
            const float e = __expf(acc[t][r] - m);
            acc[t][r] = e;
            s += e;
        }
#pragma unroll
        for (int i = 1; i < 16; i <<= 1) s += __shfl_xor(s, i, 32);
        recip[r] = 1.0f / s;                     // diagonal guarantees s >= 1
    }

    // ---- Stage unnormalized attn (bf16) in wave-private LDS ----
    __bf16* myattn = lds_attn + wave * (16 * 256);
#pragma unroll
    for (int t = 0; t < 16; ++t)
#pragma unroll
        for (int r = 0; r < 8; ++r)
            myattn[(r + 8 * half) * 256 + t * 16 + n16] = (__bf16)acc[t][r];
    // LDS ops are in-order within a wave -> no barrier needed (wave-private).

    // ---- Phase 2: out = attn @ v, 4 chunks of 128 output dims ----
    for (int dc = 0; dc < 4; ++dc) {
        v8f acc2[8] = {};
#pragma unroll
        for (int kk = 0; kk < 256; kk += 32) {
            const __bf16* ap = myattn + n16 * 256 + kk;
            v16bf a = load_a_frag(ap + 16 * half, ap + 16 * half + 8);
            int kpos = wstart - WSZ + kk + 16 * half;
            kpos = kpos < 0 ? 0 : kpos;          // weight is exactly 0 there
#pragma unroll
            for (int t = 0; t < 8; ++t) {
                const int dcol = dc * 128 + t * 16 + n16;
                const __bf16* vp = vT + dcol * MROWS + b * NN + kpos;
                acc2[t] = wmma_bf16(a, *(const v16bf*)vp, acc2[t]);
            }
        }
#pragma unroll
        for (int t = 0; t < 8; ++t) {
            const int dcol = dc * 128 + t * 16 + n16;
#pragma unroll
            for (int r = 0; r < 8; ++r) {
                const int qpos = wstart + wave * 16 + r + 8 * half;
                out[(size_t)(b * NN + qpos) * DD + dcol] = acc2[t][r] * recip[r];
            }
        }
    }
}

// ---------------------------------------------------------------------------
// Launch: convert -> QKV GEMM -> local attention.
// Workspace layout (bf16): x | w(q,k,v) | q | k | vT  (~270 MB)
// ---------------------------------------------------------------------------
extern "C" void kernel_launch(void* const* d_in, const int* in_sizes, int n_in,
                              void* d_out, int out_size, void* d_ws, size_t ws_size,
                              hipStream_t stream) {
    const float* x  = (const float*)d_in[0];
    const float* wq = (const float*)d_in[1];
    const float* bq = (const float*)d_in[2];
    const float* wk = (const float*)d_in[3];
    const float* bk = (const float*)d_in[4];
    const float* wv = (const float*)d_in[5];
    const float* bv = (const float*)d_in[6];
    float* out = (float*)d_out;

    __bf16* xb = (__bf16*)d_ws;
    __bf16* wb = xb + (size_t)MROWS * DD;
    __bf16* qb = wb + (size_t)3 * DD * DD;
    __bf16* kb = qb + (size_t)MROWS * DD;
    __bf16* vT = kb + (size_t)MROWS * DD;

    cvt_f32_bf16<<<(MROWS * DD) / 1024, 256, 0, stream>>>(x, xb);
    cvt_f32_bf16<<<(DD * DD) / 1024, 256, 0, stream>>>(wq, wb);
    cvt_f32_bf16<<<(DD * DD) / 1024, 256, 0, stream>>>(wk, wb + DD * DD);
    cvt_f32_bf16<<<(DD * DD) / 1024, 256, 0, stream>>>(wv, wb + 2 * DD * DD);

    qkv_gemm<<<dim3(MROWS / 128, DD / 256, 3), 256, 0, stream>>>(
        xb, wb, bq, bk, bv, qb, kb, vT);

    local_attn<<<dim3(NWIN, BB), 256, 0, stream>>>(qb, kb, vT, out);
}
